// EfficientAttention_4629974745160
// MI455X (gfx1250) — matrix operand order repaired and checked by reading, not compiled, not run on hardware
//
#include <hip/hip_runtime.h>
#include <hip/hip_bf16.h>

// EfficientAttention (linear attention) for MI455X / gfx1250, wave32 + WMMA +
// async global->LDS double buffering.
//
// Pass 1 (ctx_accum): K/V projection GEMMs in bf16 WMMA, accumulate
//   S[n,h,16,16] = sum_l exp(k_l) v_l^T, z[n,h,16] = sum_l exp(k_l).
// Pass 2 (attend_out): Q projection GEMM, per-head channel softmax,
//   attended = ctx^T @ q_sm, output GEMM (WMMA) + bias + residual.
//
// x tiles (16 pixels x 128 channels, f32) are DMA'd into LDS with
// global_load_async_to_lds_b32 (ASYNCcnt), double buffered so tile t+1
// streams in while tile t is consumed by WMMA.

#define HEADS 8
#define C     128
#define HW    65536
#define NB    4

typedef __attribute__((ext_vector_type(16))) __bf16 bf16x16;
typedef __attribute__((ext_vector_type(8)))  __bf16 bf16x8;
typedef __attribute__((ext_vector_type(8)))  float  f32x8;

// ---------------------------------------------------------------- async utils
static __device__ inline void wait_async_le8() {
#if __has_builtin(__builtin_amdgcn_s_wait_asynccnt)
  __builtin_amdgcn_s_wait_asynccnt(8);
#else
  asm volatile("s_wait_asynccnt 0x8" ::: "memory");
#endif
}
static __device__ inline void wait_async_0() {
#if __has_builtin(__builtin_amdgcn_s_wait_asynccnt)
  __builtin_amdgcn_s_wait_asynccnt(0);
#else
  asm volatile("s_wait_asynccnt 0x0" ::: "memory");
#endif
}

// Async-copy one 16px x 128ch f32 tile into LDS laid out [pixel][channel].
// Thread -> (pixel = tid&15, channels c0..c0+7). Global reads are 64B
// coalesced across the 16 pixel lanes; LDS writes are done by the async
// engine (no ds_store from the wave). 8 copies per thread per tile.
static __device__ inline void stage_tile_async(const float* __restrict__ xg,
                                               int px0, float* dstLds, int tid) {
  const int p  = tid & 15;
  const int c0 = (tid >> 4) * 8;
  // Low 32 bits of a generic pointer to __shared__ == wave-relative LDS offset.
  unsigned lds0 = (unsigned)(size_t)(dstLds + p * C + c0);
  const float* g = xg + (size_t)c0 * HW + px0 + p;
#pragma unroll
  for (int i = 0; i < 8; ++i) {
    unsigned l = lds0 + (unsigned)(i * 4);
    const float* ga = g + (size_t)i * HW;
    asm volatile("global_load_async_to_lds_b32 %0, %1, off"
                 :: "v"(l), "v"(ga) : "memory");
  }
}

// ------------------------------------------------------------------ fragments
// B fragment (32x16, KxN) from f32 LDS tile [pixel][channel], cvt -> bf16.
// Lanes 0-15: K=kb..kb+15 for column N=lane; lanes 16-31: K=kb+16..kb+31.
static __device__ inline bf16x16 load_b_frag_f32(const float* __restrict__ smem,
                                                 int lane, int kb) {
  const int nn   = lane & 15;
  const int koff = (lane < 16) ? 0 : 16;
  const float4* p = (const float4*)(smem + nn * C + kb + koff);
  float4 x0 = p[0], x1 = p[1], x2 = p[2], x3 = p[3];
  bf16x16 b;
  b[0]  = (__bf16)x0.x; b[1]  = (__bf16)x0.y; b[2]  = (__bf16)x0.z; b[3]  = (__bf16)x0.w;
  b[4]  = (__bf16)x1.x; b[5]  = (__bf16)x1.y; b[6]  = (__bf16)x1.z; b[7]  = (__bf16)x1.w;
  b[8]  = (__bf16)x2.x; b[9]  = (__bf16)x2.y; b[10] = (__bf16)x2.z; b[11] = (__bf16)x2.w;
  b[12] = (__bf16)x3.x; b[13] = (__bf16)x3.y; b[14] = (__bf16)x3.z; b[15] = (__bf16)x3.w;
  return b;
}

// B fragment from a bf16 LDS tile [pixel][channel] (used for attended tile).
static __device__ inline bf16x16 load_b_frag_bf16(const __bf16* __restrict__ smem,
                                                  int lane, int kb) {
  const int nn   = lane & 15;
  const int koff = (lane < 16) ? 0 : 16;
  const bf16x8* p = (const bf16x8*)(smem + nn * C + kb + koff);
  bf16x8 lo = p[0];
  bf16x8 hi = p[1];
  bf16x16 b;
#pragma unroll
  for (int i = 0; i < 8; ++i) { b[i] = lo[i]; b[8 + i] = hi[i]; }
  return b;
}

// A fragment (16x32, MxK) from global f32 weight matrix W[M][128], bf16-cast.
// Lanes 0-15 row M=rowBase+lane: K=kb..kb+15;
// lanes 16-31 row M=rowBase+lane-16: K=kb+16..kb+31.
static __device__ inline bf16x16 load_a_frag_global(const float* __restrict__ W,
                                                    int rowBase, int lane, int kb) {
  const int m    = rowBase + (lane & 15);
  const int koff = (lane < 16) ? 0 : 16;
  const float4* w0 = (const float4*)(W + (size_t)m * C + kb + koff);
  const float4* w1 = (const float4*)(W + (size_t)m * C + kb + 8 + koff);
  float4 a0 = w0[0], a1 = w0[1];
  float4 b0 = w1[0], b1 = w1[1];
  bf16x16 a;
  a[0]  = (__bf16)a0.x; a[1]  = (__bf16)a0.y; a[2]  = (__bf16)a0.z; a[3]  = (__bf16)a0.w;
  a[4]  = (__bf16)a1.x; a[5]  = (__bf16)a1.y; a[6]  = (__bf16)a1.z; a[7]  = (__bf16)a1.w;
  a[8]  = (__bf16)b0.x; a[9]  = (__bf16)b0.y; a[10] = (__bf16)b0.z; a[11] = (__bf16)b0.w;
  a[12] = (__bf16)b1.x; a[13] = (__bf16)b1.y; a[14] = (__bf16)b1.z; a[15] = (__bf16)b1.w;
  return a;
}

__global__ void zero_ws_kernel(float* __restrict__ p, int nfl) {
  int i = blockIdx.x * blockDim.x + threadIdx.x;
  if (i < nfl) p[i] = 0.0f;
}

// ------------------------- Pass 1: context accumulation -------------------------
__global__ __launch_bounds__(256) void ctx_accum_kernel(
    const float* __restrict__ x,
    const float* __restrict__ Wk, const float* __restrict__ bk,
    const float* __restrict__ Wv, const float* __restrict__ bv,
    float* __restrict__ Sacc, float* __restrict__ Zacc) {
  __shared__ alignas(16) float xs[2][16 * C];        // double-buffered f32 tiles
  __shared__ float ev[HEADS][2][16][16];             // per-wave exp(k) and v dumps

  const int tid  = threadIdx.x;
  const int lane = tid & 31;
  const int w    = tid >> 5;                          // wave id == head id
  const int wg   = blockIdx.x;
  const int ng   = wg >> 8;                           // image (256 blocks / image)
  const int pblk = (wg & 255) * 256;                  // 256 pixels per block
  const int mh   = (lane >> 4) * 8;                   // C-layout row offset
  const int col  = lane & 15;

  const float* xg = x + (size_t)ng * C * HW;

  // Kick off tile 0 DMA before doing anything else.
  stage_tile_async(xg, pblk, xs[0], tid);

  // Resident weight fragments (rows 16w..16w+15 of Wk and Wv), f32 -> bf16.
  bf16x16 aK[4], aV[4];
#pragma unroll
  for (int ks = 0; ks < 4; ++ks) {
    aK[ks] = load_a_frag_global(Wk, w * 16, lane, ks * 32);
    aV[ks] = load_a_frag_global(Wv, w * 16, lane, ks * 32);
  }
  float bkv[8], bvv[8];
#pragma unroll
  for (int j = 0; j < 8; ++j) {
    int r = w * 16 + j + mh;
    bkv[j] = bk[r];
    bvv[j] = bv[r];
  }

  float Sloc[8];
#pragma unroll
  for (int j = 0; j < 8; ++j) Sloc[j] = 0.0f;
  float zloc = 0.0f;

  for (int t = 0; t < 16; ++t) {
    const float* buf = xs[t & 1];
    // Prefetch next tile into the other buffer, then drain this tile's copies.
    if (t + 1 < 16) {
      stage_tile_async(xg, pblk + (t + 1) * 16, xs[(t + 1) & 1], tid);
      wait_async_le8();
    } else {
      wait_async_0();
    }
    __syncthreads();                                  // tile t present for all waves

    f32x8 kacc = {0.f, 0.f, 0.f, 0.f, 0.f, 0.f, 0.f, 0.f};
    f32x8 vacc = {0.f, 0.f, 0.f, 0.f, 0.f, 0.f, 0.f, 0.f};
#pragma unroll
    for (int ks = 0; ks < 4; ++ks) {
      bf16x16 b = load_b_frag_f32(buf, lane, ks * 32);
      kacc = __builtin_amdgcn_wmma_f32_16x16x32_bf16(false, aK[ks], false, b,
                                                     (short)0, kacc, false, false);
      vacc = __builtin_amdgcn_wmma_f32_16x16x32_bf16(false, aV[ks], false, b,
                                                     (short)0, vacc, false, false);
    }

#pragma unroll
    for (int j = 0; j < 8; ++j) {
      int m = j + mh;
      ev[w][0][m][col] = __expf(kacc[j] + bkv[j]);
      ev[w][1][m][col] = vacc[j] + bvv[j];
    }
    // Barrier: (a) ev visible wave-wide, (b) every wave is done reading buf,
    // so the next iteration may issue DMA that overwrites it.
    __syncthreads();

    if (lane < 16) {
      float s = 0.0f;
#pragma unroll
      for (int p = 0; p < 16; ++p) s += ev[w][0][lane][p];
      zloc += s;
    }
#pragma unroll
    for (int j = 0; j < 8; ++j) {
      int idx = lane * 8 + j;
      int kr = idx >> 4, vr = idx & 15;
      float s = 0.0f;
#pragma unroll
      for (int p = 0; p < 16; ++p) s += ev[w][0][kr][p] * ev[w][1][vr][p];
      Sloc[j] += s;
    }
  }

  float* Sg = Sacc + (size_t)(ng * HEADS + w) * 256;
  float* Zg = Zacc + (size_t)(ng * HEADS + w) * 16;
#pragma unroll
  for (int j = 0; j < 8; ++j) atomicAdd(Sg + lane * 8 + j, Sloc[j]);
  if (lane < 16) atomicAdd(Zg + lane, zloc);
}

// ------------------------- Pass 2: attend + output GEMM -------------------------
__global__ __launch_bounds__(256) void attend_out_kernel(
    const float* __restrict__ x,
    const float* __restrict__ Wq, const float* __restrict__ bq,
    const float* __restrict__ Wr, const float* __restrict__ br,
    const float* __restrict__ Sacc, const float* __restrict__ Zacc,
    float* __restrict__ out) {
  __shared__ alignas(16) float  xs[2][16 * C];        // double-buffered f32 tiles
  __shared__ alignas(16) __bf16 att[16 * C];          // [pixel][v-channel]
  __shared__ float ctx[HEADS][16][16];                // context[k][v] per head
  __shared__ float qsm[HEADS][16][16];                // q-softmax per wave

  const int tid  = threadIdx.x;
  const int lane = tid & 31;
  const int w    = tid >> 5;
  const int wg   = blockIdx.x;
  const int ng   = wg >> 8;
  const int pblk = (wg & 255) * 256;
  const int mh   = (lane >> 4) * 8;
  const int col  = lane & 15;

  const float* xg = x + (size_t)ng * C * HW;
  float*       og = out + (size_t)ng * C * HW;

  stage_tile_async(xg, pblk, xs[0], tid);

  // Finalize context = S / z for this image (2048 entries / 256 threads).
  {
    const float* Sg = Sacc + (size_t)ng * HEADS * 256;
    const float* Zg = Zacc + (size_t)ng * HEADS * 16;
    for (int i = tid; i < HEADS * 256; i += 256) {
      int h = i >> 8, rem = i & 255, k = rem >> 4, v = rem & 15;
      ctx[h][k][v] = Sg[i] / Zg[h * 16 + k];
    }
  }

  bf16x16 aQ[4], aR[4];
#pragma unroll
  for (int ks = 0; ks < 4; ++ks) {
    aQ[ks] = load_a_frag_global(Wq, w * 16, lane, ks * 32);
    aR[ks] = load_a_frag_global(Wr, w * 16, lane, ks * 32);
  }
  float bqv[8], brv[8];
#pragma unroll
  for (int j = 0; j < 8; ++j) {
    int r = w * 16 + j + mh;
    bqv[j] = bq[r];
    brv[j] = br[r];
  }
  __syncthreads();                                    // ctx visible

  for (int t = 0; t < 16; ++t) {
    const int px0 = pblk + t * 16;
    const float* buf = xs[t & 1];
    if (t + 1 < 16) {
      stage_tile_async(xg, pblk + (t + 1) * 16, xs[(t + 1) & 1], tid);
      wait_async_le8();
    } else {
      wait_async_0();
    }
    __syncthreads();                                  // tile t present

    // Q projection for head w.
    f32x8 qacc = {0.f, 0.f, 0.f, 0.f, 0.f, 0.f, 0.f, 0.f};
#pragma unroll
    for (int ks = 0; ks < 4; ++ks) {
      bf16x16 b = load_b_frag_f32(buf, lane, ks * 32);
      qacc = __builtin_amdgcn_wmma_f32_16x16x32_bf16(false, aQ[ks], false, b,
                                                     (short)0, qacc, false, false);
    }

    // Softmax over the 16 head channels (column-wise in C layout).
    float e[8];
    float s = 0.0f;
#pragma unroll
    for (int j = 0; j < 8; ++j) { e[j] = __expf(qacc[j] + bqv[j]); s += e[j]; }
    s += __shfl_xor(s, 16, 32);
    float inv = 1.0f / s;
#pragma unroll
    for (int j = 0; j < 8; ++j) qsm[w][j + mh][col] = e[j] * inv;
    __builtin_amdgcn_wave_barrier();                  // same-wave LDS RAW ordering

    // attended[v][p] = sum_k ctx[k][v] * qsm[k][p]; lane owns 8 entries.
#pragma unroll
    for (int j = 0; j < 8; ++j) {
      int idx = lane * 8 + j;
      int v = idx >> 4, p = idx & 15;
      float a = 0.0f;
#pragma unroll
      for (int k = 0; k < 16; ++k) a += ctx[w][k][v] * qsm[w][k][p];
      att[p * C + w * 16 + v] = (__bf16)a;
    }
    // Barrier: (a) att assembled across waves, (b) all xs reads of tile t done,
    // so next iteration's DMA may overwrite the buffer.
    __syncthreads();

    // Output projection rows 16w..16w+15 over all 128 v-channels.
    f32x8 racc = {0.f, 0.f, 0.f, 0.f, 0.f, 0.f, 0.f, 0.f};
#pragma unroll
    for (int ks = 0; ks < 4; ++ks) {
      bf16x16 b = load_b_frag_bf16(att, lane, ks * 32);
      racc = __builtin_amdgcn_wmma_f32_16x16x32_bf16(false, aR[ks], false, b,
                                                     (short)0, racc, false, false);
    }

    // Epilogue: + br + residual x, store.
#pragma unroll
    for (int j = 0; j < 8; ++j) {
      int row = w * 16 + j + mh;
      size_t off = (size_t)row * HW + px0 + col;
      og[off] = racc[j] + brv[j] + xg[off];
    }
  }
}

extern "C" void kernel_launch(void* const* d_in, const int* in_sizes, int n_in,
                              void* d_out, int out_size, void* d_ws, size_t ws_size,
                              hipStream_t stream) {
  (void)in_sizes; (void)n_in; (void)out_size; (void)ws_size;
  const float* x  = (const float*)d_in[0];
  const float* Wk = (const float*)d_in[1];
  const float* bk = (const float*)d_in[2];
  const float* Wq = (const float*)d_in[3];
  const float* bq = (const float*)d_in[4];
  const float* Wv = (const float*)d_in[5];
  const float* bv = (const float*)d_in[6];
  const float* Wr = (const float*)d_in[7];
  const float* br = (const float*)d_in[8];
  float* out = (float*)d_out;

  float* Sacc = (float*)d_ws;                         // [4][8][16][16]
  float* Zacc = Sacc + NB * HEADS * 256;              // [4][8][16]
  const int nacc = NB * HEADS * 256 + NB * HEADS * 16;

  zero_ws_kernel<<<(nacc + 255) / 256, 256, 0, stream>>>(Sacc, nacc);
  ctx_accum_kernel<<<NB * 256, 256, 0, stream>>>(x, Wk, bk, Wv, bv, Sacc, Zacc);
  attend_out_kernel<<<NB * 256, 256, 0, stream>>>(x, Wq, bq, Wr, br, Sacc, Zacc, out);
}
